// LinearMemoryAttention_28956669509840
// MI455X (gfx1250) — matrix operand order repaired and checked by reading, not compiled, not run on hardware
//
#include <hip/hip_runtime.h>
#include <hip/hip_bf16.h>

// ---------------------------------------------------------------------------
// LinearMemoryAttention for MI455X (gfx1250), wave32.
//   out = ( l2norm(H @ Wq + bq) per-head @ M0 ) @ Wo     (K/V paths are dead)
// bf16 WMMA (v_wmma_f32_16x16x32_bf16) + async global->LDS staging +
// double-buffered K loop.  B=4,S=8192,HID=PROJ=1024,NH=16,HD=64, M=32768.
// ---------------------------------------------------------------------------

typedef __attribute__((ext_vector_type(16))) __bf16 v16bf;
typedef __attribute__((ext_vector_type(8)))  float  v8f;
typedef int vi4 __attribute__((vector_size(16)));   // matches builtin param type

#define M_TOT   32768
#define DIM     1024
#define NHEAD   16
#define HDIM    64

#if defined(__HIP_DEVICE_COMPILE__) && __has_builtin(__builtin_amdgcn_global_load_async_to_lds_b128)
#define HAS_ASYNC_LDS 1
#else
#define HAS_ASYNC_LDS 0
#endif

// ---- async global->LDS b128 copy (CDNA5 GLOBAL_LOAD_ASYNC_TO_LDS_B128) ----
__device__ __forceinline__ void async_b128(const void* g, void* l) {
#if HAS_ASYNC_LDS
    __builtin_amdgcn_global_load_async_to_lds_b128(
        (__attribute__((address_space(1))) vi4*)(g),
        (__attribute__((address_space(3))) vi4*)(l),
        0, 0);
#else
    *reinterpret_cast<uint4*>(l) = *reinterpret_cast<const uint4*>(g);
#endif
}

__device__ __forceinline__ void wait_async() {
#if HAS_ASYNC_LDS
#if __has_builtin(__builtin_amdgcn_s_wait_asynccnt)
    __builtin_amdgcn_s_wait_asynccnt(0);
#else
    asm volatile("s_wait_asynccnt 0x0" ::: "memory");
#endif
#endif
}

// ---- WMMA wrapper ----------------------------------------------------------
__device__ __forceinline__ v8f wmma_bf16(v16bf a, v16bf b, v8f c) {
    return __builtin_amdgcn_wmma_f32_16x16x32_bf16(false, a, false, b,
                                                   (short)0, c, false, false);
}

// ---- fragment loaders (wave32 layouts per CDNA5 ISA 7.12.2) ---------------
// A (16x32): lanes 0-15 M=lane  K[0..7]+K[16..23]; lanes16-31 K[8..15]+K[24..31]
__device__ __forceinline__ v16bf ld_frag_a(const __bf16* lds, int stride,
                                           int row0, int lane) {
    const int half = lane >> 4, lr = lane & 15;
    const __bf16* p = lds + (row0 + lr) * stride + half * 8;
    union { v16bf v; uint4 q[2]; } u;
    u.q[0] = *reinterpret_cast<const uint4*>(p);
    u.q[1] = *reinterpret_cast<const uint4*>(p + 16);
    return u.v;
}
// B (32x16): staged transposed [n][k]; lanes 0-15 K[0..7]+K[16..23]; lanes16-31 K[8..15]+K[24..31]
__device__ __forceinline__ v16bf ld_frag_b(const __bf16* lds, int stride,
                                           int col0, int lane) {
    const int half = lane >> 4, lr = lane & 15;
    const __bf16* p = lds + (col0 + lr) * stride + half * 8;
    union { v16bf v; uint4 q[2]; } u;
    u.q[0] = *reinterpret_cast<const uint4*>(p);
    u.q[1] = *reinterpret_cast<const uint4*>(p + 16);
    return u.v;
}

// ===========================================================================
// Prep: batched transpose + f32->bf16:  out[bz][c][r] = bf16(in[bz][r][c])
// block 256 = 32x8, 32x32 tiles. Used for WqT, WoT, M0T (tiny traffic).
// ===========================================================================
__global__ __launch_bounds__(256)
void lma_transpose_cvt_kernel(const float* __restrict__ in,
                              __bf16* __restrict__ out, int R, int C) {
    __shared__ __bf16 t[32 * 33];
    const int tid = threadIdx.x, tx = tid & 31, ty = tid >> 5;
    const int r0 = blockIdx.y * 32, c0 = blockIdx.x * 32;
    const size_t base = (size_t)blockIdx.z * R * C;
#pragma unroll
    for (int j = 0; j < 4; ++j) {
        const int r = ty + j * 8;
        t[r * 33 + tx] = (__bf16)in[base + (size_t)(r0 + r) * C + c0 + tx];
    }
    __syncthreads();
#pragma unroll
    for (int j = 0; j < 4; ++j) {
        const int c = ty + j * 8;
        out[base + (size_t)(c0 + c) * R + r0 + tx] = t[tx * 33 + c];
    }
}

// ===========================================================================
// Kernel 1: P = bf16( H @ Wq + bq ), rowSumSq += per-row sum of squares.
// grid (DIM/64, M_TOT/128), 8 waves, wave = 32x32 region. Double-buffered.
// A: f32->bf16 via VGPRs (H is f32). B: async copy from pre-transposed WqT.
// ===========================================================================
__global__ __launch_bounds__(256)
void lma_qproj_kernel(const float* __restrict__ H,
                      const __bf16* __restrict__ WqT,   // [n][k] bf16
                      const float* __restrict__ bq,
                      __bf16* __restrict__ P,
                      float* __restrict__ rowSumSq) {
    __shared__ __align__(16) __bf16 As[2][128 * 40];
    __shared__ __align__(16) __bf16 Bs[2][64 * 40];
    __shared__ float rowsq[128];

    const int tid  = threadIdx.x;
    const int lane = tid & 31, wave = tid >> 5;
    const int wm   = wave >> 1, wn = wave & 1;
    const int half = lane >> 4, lr = lane & 15;
    const int m0 = blockIdx.y * 128, n0 = blockIdx.x * 64;

    if (tid < 128) rowsq[tid] = 0.0f;

    const int bn = tid >> 2, bk8 = (tid & 3) << 3;       // B-tile async coords

    // ---- prologue: stage tile 0 into buffer 0 ----
    async_b128(WqT + (size_t)(n0 + bn) * DIM + 0 + bk8, &Bs[0][bn * 40 + bk8]);
    {
        float4 w[4];
#pragma unroll
        for (int i = 0; i < 4; ++i) {
            const int idx = tid + i * 256;
            const int row = idx >> 3, c4 = (idx & 7) << 2;
            w[i] = *reinterpret_cast<const float4*>(
                H + (size_t)(m0 + row) * DIM + c4);
        }
#pragma unroll
        for (int i = 0; i < 4; ++i) {
            const int idx = tid + i * 256;
            const int row = idx >> 3, c4 = (idx & 7) << 2;
            union { __bf16 h[4]; uint2 u; } t;
            t.h[0] = (__bf16)w[i].x; t.h[1] = (__bf16)w[i].y;
            t.h[2] = (__bf16)w[i].z; t.h[3] = (__bf16)w[i].w;
            *reinterpret_cast<uint2*>(&As[0][row * 40 + c4]) = t.u;
        }
    }
    wait_async();
    __syncthreads();

    const v8f vz = {0.f,0.f,0.f,0.f,0.f,0.f,0.f,0.f};
    v8f acc[2][2] = {{vz, vz}, {vz, vz}};

    const int NS = DIM / 32;
    for (int s = 0; s < NS; ++s) {
        const int cur = s & 1, nxt = cur ^ 1;
        const int kn  = (s + 1) * 32;
        float4 w[4];
        if (s + 1 < NS) {
            // issue next B async + next A global loads before compute
            async_b128(WqT + (size_t)(n0 + bn) * DIM + kn + bk8,
                       &Bs[nxt][bn * 40 + bk8]);
#pragma unroll
            for (int i = 0; i < 4; ++i) {
                const int idx = tid + i * 256;
                const int row = idx >> 3, c4 = (idx & 7) << 2;
                w[i] = *reinterpret_cast<const float4*>(
                    H + (size_t)(m0 + row) * DIM + kn + c4);
            }
        }
        // compute from current buffers
        v16bf af[2], bfr[2];
#pragma unroll
        for (int mi = 0; mi < 2; ++mi)
            af[mi] = ld_frag_a(As[cur], 40, wm * 32 + mi * 16, lane);
#pragma unroll
        for (int ni = 0; ni < 2; ++ni)
            bfr[ni] = ld_frag_b(Bs[cur], 40, wn * 32 + ni * 16, lane);
#pragma unroll
        for (int mi = 0; mi < 2; ++mi)
#pragma unroll
            for (int ni = 0; ni < 2; ++ni)
                acc[mi][ni] = wmma_bf16(af[mi], bfr[ni], acc[mi][ni]);
        if (s + 1 < NS) {
#pragma unroll
            for (int i = 0; i < 4; ++i) {
                const int idx = tid + i * 256;
                const int row = idx >> 3, c4 = (idx & 7) << 2;
                union { __bf16 h[4]; uint2 u; } t;
                t.h[0] = (__bf16)w[i].x; t.h[1] = (__bf16)w[i].y;
                t.h[2] = (__bf16)w[i].z; t.h[3] = (__bf16)w[i].w;
                *reinterpret_cast<uint2*>(&As[nxt][row * 40 + c4]) = t.u;
            }
        }
        wait_async();
        __syncthreads();
    }

    // epilogue: bias, bf16 store of P, per-row sum of squares
    float sq[2][8] = {};
#pragma unroll
    for (int mi = 0; mi < 2; ++mi) {
#pragma unroll
        for (int ni = 0; ni < 2; ++ni) {
            const int gn   = n0 + wn * 32 + ni * 16 + lr;
            const float bs = bq[gn];
#pragma unroll
            for (int r = 0; r < 8; ++r) {
                const int   rowl = wm * 32 + mi * 16 + half * 8 + r;
                const float x    = acc[mi][ni][r] + bs;
                P[(size_t)(m0 + rowl) * DIM + gn] = (__bf16)x;
                sq[mi][r] += x * x;
            }
        }
    }
#pragma unroll
    for (int mi = 0; mi < 2; ++mi)
#pragma unroll
        for (int r = 0; r < 8; ++r)
            atomicAdd(&rowsq[wm * 32 + mi * 16 + half * 8 + r], sq[mi][r]);
    __syncthreads();
    if (tid < 128) atomicAdd(&rowSumSq[m0 + tid], rowsq[tid]);
}

// ===========================================================================
// Kernel 2: R[r,h*64+d] = (1/max(||P_row||,1e-12)) * sum_e P[r,h*64+e]*M0[b,h,e,d]
// grid (NHEAD, M_TOT/128). K=64 staged once, fully async.
// ===========================================================================
__global__ __launch_bounds__(256)
void lma_retrieve_kernel(const __bf16* __restrict__ P,
                         const __bf16* __restrict__ M0T,  // [b][h][d][e] bf16
                         const float*  __restrict__ rowSumSq,
                         __bf16* __restrict__ R) {
    __shared__ __align__(16) __bf16 As[128 * 72];
    __shared__ __align__(16) __bf16 Bs[64 * 72];
    __shared__ float rscale[128];

    const int tid  = threadIdx.x;
    const int lane = tid & 31, wave = tid >> 5;
    const int wm   = wave >> 1, wn = wave & 1;
    const int half = lane >> 4, lr = lane & 15;
    const int h  = blockIdx.x;
    const int m0 = blockIdx.y * 128;
    const int b  = m0 >> 13;

    // stage A: 128x64 bf16 async
#pragma unroll
    for (int i = 0; i < 4; ++i) {
        const int idx = tid + i * 256;
        const int row = idx >> 3, c8 = (idx & 7) << 3;
        async_b128(P + (size_t)(m0 + row) * DIM + h * HDIM + c8,
                   &As[row * 72 + c8]);
    }
    // stage B: M0T[b][h] 64(d) x 64(e) bf16 async
#pragma unroll
    for (int i = 0; i < 2; ++i) {
        const int idx = tid + i * 256;
        const int d = idx >> 3, e8 = (idx & 7) << 3;
        async_b128(M0T + (((size_t)(b * NHEAD + h) * HDIM + d) * HDIM) + e8,
                   &Bs[d * 72 + e8]);
    }
    if (tid < 128) {
        const float ss = rowSumSq[m0 + tid];
        rscale[tid] = 1.0f / fmaxf(sqrtf(ss), 1e-12f);
    }
    wait_async();
    __syncthreads();

    const v8f vz = {0.f,0.f,0.f,0.f,0.f,0.f,0.f,0.f};
    v8f acc[2][2] = {{vz, vz}, {vz, vz}};

#pragma unroll
    for (int kk = 0; kk < HDIM; kk += 32) {
        v16bf af[2], bfr[2];
#pragma unroll
        for (int mi = 0; mi < 2; ++mi)
            af[mi] = ld_frag_a(As + kk, 72, wm * 32 + mi * 16, lane);
#pragma unroll
        for (int ni = 0; ni < 2; ++ni)
            bfr[ni] = ld_frag_b(Bs + kk, 72, wn * 32 + ni * 16, lane);
#pragma unroll
        for (int mi = 0; mi < 2; ++mi)
#pragma unroll
            for (int ni = 0; ni < 2; ++ni)
                acc[mi][ni] = wmma_bf16(af[mi], bfr[ni], acc[mi][ni]);
    }

#pragma unroll
    for (int mi = 0; mi < 2; ++mi) {
#pragma unroll
        for (int ni = 0; ni < 2; ++ni) {
            const int gn = h * HDIM + wn * 32 + ni * 16 + lr;
#pragma unroll
            for (int r = 0; r < 8; ++r) {
                const int   rowl = wm * 32 + mi * 16 + half * 8 + r;
                const float x    = acc[mi][ni][r] * rscale[rowl];
                R[(size_t)(m0 + rowl) * DIM + gn] = (__bf16)x;
            }
        }
    }
}

// ===========================================================================
// Kernel 3: out(f32) = R(bf16) @ Wo. Fully async staging, double-buffered.
// ===========================================================================
__global__ __launch_bounds__(256)
void lma_oproj_kernel(const __bf16* __restrict__ R,
                      const __bf16* __restrict__ WoT,   // [n][k] bf16
                      float* __restrict__ OUT) {
    __shared__ __align__(16) __bf16 As[2][128 * 40];
    __shared__ __align__(16) __bf16 Bs[2][64 * 40];

    const int tid  = threadIdx.x;
    const int lane = tid & 31, wave = tid >> 5;
    const int wm   = wave >> 1, wn = wave & 1;
    const int half = lane >> 4, lr = lane & 15;
    const int m0 = blockIdx.y * 128, n0 = blockIdx.x * 64;

    const int bn = tid >> 2, bk8 = (tid & 3) << 3;

    // prologue: stage tile 0
#pragma unroll
    for (int i = 0; i < 2; ++i) {
        const int idx = tid + i * 256;
        const int row = idx >> 2, k8 = (idx & 3) << 3;
        async_b128(R + (size_t)(m0 + row) * DIM + k8, &As[0][row * 40 + k8]);
    }
    async_b128(WoT + (size_t)(n0 + bn) * DIM + bk8, &Bs[0][bn * 40 + bk8]);
    wait_async();
    __syncthreads();

    const v8f vz = {0.f,0.f,0.f,0.f,0.f,0.f,0.f,0.f};
    v8f acc[2][2] = {{vz, vz}, {vz, vz}};

    const int NS = DIM / 32;
    for (int s = 0; s < NS; ++s) {
        const int cur = s & 1, nxt = cur ^ 1;
        const int kn  = (s + 1) * 32;
        if (s + 1 < NS) {
#pragma unroll
            for (int i = 0; i < 2; ++i) {
                const int idx = tid + i * 256;
                const int row = idx >> 2, k8 = (idx & 3) << 3;
                async_b128(R + (size_t)(m0 + row) * DIM + kn + k8,
                           &As[nxt][row * 40 + k8]);
            }
            async_b128(WoT + (size_t)(n0 + bn) * DIM + kn + bk8,
                       &Bs[nxt][bn * 40 + bk8]);
        }
        v16bf af[2], bfr[2];
#pragma unroll
        for (int mi = 0; mi < 2; ++mi)
            af[mi] = ld_frag_a(As[cur], 40, wm * 32 + mi * 16, lane);
#pragma unroll
        for (int ni = 0; ni < 2; ++ni)
            bfr[ni] = ld_frag_b(Bs[cur], 40, wn * 32 + ni * 16, lane);
#pragma unroll
        for (int mi = 0; mi < 2; ++mi)
#pragma unroll
            for (int ni = 0; ni < 2; ++ni)
                acc[mi][ni] = wmma_bf16(af[mi], bfr[ni], acc[mi][ni]);
        wait_async();
        __syncthreads();
    }

#pragma unroll
    for (int mi = 0; mi < 2; ++mi) {
#pragma unroll
        for (int ni = 0; ni < 2; ++ni) {
            const int gn = n0 + wn * 32 + ni * 16 + lr;
#pragma unroll
            for (int r = 0; r < 8; ++r) {
                const int rowl = wm * 32 + mi * 16 + half * 8 + r;
                OUT[(size_t)(m0 + rowl) * DIM + gn] = acc[mi][ni][r];
            }
        }
    }
}

// ===========================================================================
extern "C" void kernel_launch(void* const* d_in, const int* in_sizes, int n_in,
                              void* d_out, int out_size, void* d_ws, size_t ws_size,
                              hipStream_t stream) {
    (void)in_sizes; (void)n_in; (void)out_size; (void)ws_size;

    const float* H  = (const float*)d_in[0];  // [B,S,HID]
    const float* Wq = (const float*)d_in[1];  // [HID,PROJ]
    const float* bq = (const float*)d_in[2];  // [PROJ]
    // d_in[3..6] = Wk,bk,Wv,bv: dead w.r.t. the returned output
    const float* Wo = (const float*)d_in[7];  // [PROJ,HID]
    const float* M0 = (const float*)d_in[8];  // [B,NH,HD,HD]
    float* OUT = (float*)d_out;

    // workspace layout
    const size_t PB = (size_t)M_TOT * DIM * sizeof(__bf16);        // 64 MB
    const size_t WB = (size_t)DIM * DIM * sizeof(__bf16);          // 2 MB
    char*   ws   = (char*)d_ws;
    __bf16* P    = (__bf16*)(ws);
    __bf16* Rn   = (__bf16*)(ws + PB);
    float*  rss  = (float*) (ws + 2 * PB);
    __bf16* WqT  = (__bf16*)(ws + 2 * PB + (1 << 20));
    __bf16* WoT  = (__bf16*)(ws + 2 * PB + (1 << 20) + WB);
    __bf16* M0T  = (__bf16*)(ws + 2 * PB + (1 << 20) + 2 * WB);

    (void)hipMemsetAsync(rss, 0, (size_t)M_TOT * sizeof(float), stream);

    dim3 blk(256);
    // weight / memory-state transpose+convert (bf16, B-operand friendly)
    lma_transpose_cvt_kernel<<<dim3(DIM / 32, DIM / 32, 1), blk, 0, stream>>>(
        Wq, WqT, DIM, DIM);
    lma_transpose_cvt_kernel<<<dim3(DIM / 32, DIM / 32, 1), blk, 0, stream>>>(
        Wo, WoT, DIM, DIM);
    lma_transpose_cvt_kernel<<<dim3(2, 2, 4 * NHEAD), blk, 0, stream>>>(
        M0, M0T, HDIM, HDIM);

    lma_qproj_kernel<<<dim3(DIM / 64, M_TOT / 128), blk, 0, stream>>>(
        H, WqT, bq, P, rss);
    lma_retrieve_kernel<<<dim3(NHEAD, M_TOT / 128), blk, 0, stream>>>(
        P, M0T, rss, Rn);
    lma_oproj_kernel<<<dim3(DIM / 64, M_TOT / 128), blk, 0, stream>>>(
        Rn, WoT, OUT);
}
